// Attention_40175124087420
// MI455X (gfx1250) — matrix operand order repaired and checked by reading, not compiled, not run on hardware
//
#include <hip/hip_runtime.h>
#include <hip/hip_bf16.h>

#define B_  8
#define C_  256
#define HW_ 4096

typedef __attribute__((ext_vector_type(16))) __bf16 v16bf;
typedef __attribute__((ext_vector_type(8)))  __bf16 v8bf;
typedef __attribute__((ext_vector_type(8)))  float  v8f;
typedef __attribute__((ext_vector_type(4)))  float  v4f;

static __device__ inline v16bf join8(v8bf lo, v8bf hi) {
  v16bf r;
#pragma unroll
  for (int i = 0; i < 8; ++i) { r[i] = lo[i]; r[i + 8] = hi[i]; }
  return r;
}

static __device__ inline v8f wmma_bf16(v16bf a, v16bf b, v8f c) {
  // 8 args: (neg_a, A, neg_b, B, c_mod, C, reuse_a, reuse_b)
  return __builtin_amdgcn_wmma_f32_16x16x32_bf16(false, a, false, b, (short)0, c,
                                                 false, false);
}

// ---------------------------------------------------------------------------
// Kernel 0: convert the three 256x256 f32 weight matrices to bf16 (row-major,
// (out_c, in_c) as in the reference; wq.T access pattern is contiguous in in_c)
// ---------------------------------------------------------------------------
__global__ void cvt_weights(const float* __restrict__ wq,
                            const float* __restrict__ wk,
                            const float* __restrict__ wv,
                            __bf16* __restrict__ wb) {
  int idx = blockIdx.x * blockDim.x + threadIdx.x;
  if (idx >= 3 * C_ * C_) return;
  const float* src = (idx < C_ * C_) ? wq : (idx < 2 * C_ * C_ ? wk : wv);
  int off = idx & (C_ * C_ - 1);
  wb[idx] = (__bf16)src[off];
}

// ---------------------------------------------------------------------------
// Kernel 1: QKV projection with bf16 WMMA.
//   grid = (HW/64, B), block = 128 (4 waves). Each WG: 64 pixels x 256 chans.
//   q, k stored (b, n, c) bf16;  v stored transposed (b, c, n) bf16.
// ---------------------------------------------------------------------------
__global__ __launch_bounds__(128) void qkv_proj(
    const float* __restrict__ x, const __bf16* __restrict__ wb,
    const float* __restrict__ bq, const float* __restrict__ bk,
    const float* __restrict__ bv,
    __bf16* __restrict__ qb, __bf16* __restrict__ kb, __bf16* __restrict__ vt) {
  __shared__ __bf16 xs[64 * C_];  // 32 KB: x tile, pixel-major (xs[i][c])
  const int b  = blockIdx.y;
  const int n0 = blockIdx.x * 64;
  const int tid = threadIdx.x;
  const float* xb = x + (size_t)b * C_ * HW_;

  // load x[b][c][n0..n0+63], transpose into LDS as bf16
#pragma unroll 4
  for (int e = tid; e < 64 * C_; e += 128) {
    int c = e >> 6, i = e & 63;
    xs[i * C_ + c] = (__bf16)xb[(size_t)c * HW_ + n0 + i];
  }
  __syncthreads();

  const int wave = tid >> 5, lane = tid & 31;
  const int m0    = wave * 16;
  const int row16 = lane & 15;
  const int half  = lane >> 4;
  const int kb8   = half * 8;   // unused; A fragment also uses kb16
  const int kb16  = half * 16;  // B-fragment K base

  // A fragments: 16 pixels x 256 channels (8 fragments of 16x32)
  v16bf afrag[8];
#pragma unroll
  for (int f = 0; f < 8; ++f) {
    const __bf16* p = &xs[(m0 + row16) * C_ + f * 32];
    afrag[f] = *(const v16bf*)(p + kb16);
  }

  const float* biases[3] = {bq, bk, bv};
  for (int mtx = 0; mtx < 3; ++mtx) {
    const __bf16* w    = wb + (size_t)mtx * C_ * C_;
    const float*  bias = biases[mtx];
    for (int ct = 0; ct < 16; ++ct) {
      const int co = ct * 16 + row16;
      const float bvv = bias[co];
      v8f acc;
#pragma unroll
      for (int r = 0; r < 8; ++r) acc[r] = bvv;  // bias per output column
#pragma unroll
      for (int f = 0; f < 8; ++f) {
        v16bf bfrag = *(const v16bf*)(w + (size_t)co * C_ + f * 32 + kb16);
        acc = wmma_bf16(afrag[f], bfrag, acc);
      }
      if (mtx < 2) {
        __bf16* dst = (mtx == 0) ? qb : kb;
#pragma unroll
        for (int r = 0; r < 8; ++r) {
          int m = m0 + r + 8 * half;
          dst[((size_t)b * HW_ + n0 + m) * C_ + co] = (__bf16)acc[r];
        }
      } else {  // V transposed: (b, c, n)
#pragma unroll
        for (int r = 0; r < 8; ++r) {
          int m = m0 + r + 8 * half;
          vt[((size_t)b * C_ + co) * HW_ + n0 + m] = (__bf16)acc[r];
        }
      }
    }
  }
}

// ---------------------------------------------------------------------------
// Kernel 2: flash attention, M=32 per wave (two stacked 16-row A tiles).
//   grid = (HW/128, B), block = 128 (4 waves) -> 128 query rows per WG.
//   Every K/V B-fragment load feeds TWO wmmas (0.5 loads/wmma), halving the
//   L1 bandwidth per FLOP vs M=16. ~530 VGPRs/lane -> 1 wave/SIMD, by design
//   (__launch_bounds__(128,1)); watch for scratch ops = spills.
// ---------------------------------------------------------------------------
__global__ __launch_bounds__(128, 1) void flash_attn(
    const __bf16* __restrict__ qb, const __bf16* __restrict__ kb,
    const __bf16* __restrict__ vt, float* __restrict__ out) {
  __shared__ __bf16 plds[4 * 32 * 64];  // 16 KB: per-wave P staging (32x64)
  const int b   = blockIdx.y;
  const int q0  = blockIdx.x * 128;
  const int tid = threadIdx.x;
  const int wave = tid >> 5, lane = tid & 31;
  const int row16 = lane & 15;
  const int half  = lane >> 4;
  const int kb8   = half * 8;
  const int kb16  = half * 16;
  const int qrow0 = q0 + wave * 32;

  const __bf16* qbb = qb + (size_t)b * HW_ * C_;
  const __bf16* kbb = kb + (size_t)b * HW_ * C_;
  const __bf16* vtb = vt + (size_t)b * C_ * HW_;

  // Q fragments: 2 row-tiles x 256 channels, resident (128 VGPRs)
  v16bf qf[2][8];
#pragma unroll
  for (int a = 0; a < 2; ++a)
#pragma unroll
    for (int f = 0; f < 8; ++f) {
      const __bf16* p = qbb + (size_t)(qrow0 + a * 16 + row16) * C_ + f * 32;
      qf[a][f] = *(const v16bf*)(p + kb16);
    }

  v8f oacc[2][16];  // 32 rows x 256 channels f32 accumulator (256 VGPRs)
#pragma unroll
  for (int a = 0; a < 2; ++a)
#pragma unroll
    for (int ct = 0; ct < 16; ++ct)
#pragma unroll
      for (int r = 0; r < 8; ++r) oacc[a][ct][r] = 0.f;

  float mrow[2][8], lrow[2][8];
#pragma unroll
  for (int a = 0; a < 2; ++a)
#pragma unroll
    for (int r = 0; r < 8; ++r) { mrow[a][r] = -3.0e38f; lrow[a][r] = 0.f; }

  __bf16* myp = plds + wave * (32 * 64);
  const float sc = 0.0625f;  // 1/sqrt(256)

  for (int j0 = 0; j0 < HW_; j0 += 64) {
    if (j0 + 64 < HW_) {  // -> global_prefetch_b8 for next K tile
      __builtin_prefetch(kbb + (size_t)(j0 + 64 + row16) * C_, 0, 1);
      __builtin_prefetch(vtb + (size_t)row16 * HW_ + j0 + 64, 0, 1);
    }

    // S = Q K^T : 2x4 tiles of 16x16, K-dim 256; B fragment shared by a=0,1
    v8f s[2][4];
#pragma unroll
    for (int a = 0; a < 2; ++a)
#pragma unroll
      for (int nt = 0; nt < 4; ++nt)
#pragma unroll
        for (int r = 0; r < 8; ++r) s[a][nt][r] = 0.f;
#pragma unroll
    for (int nt = 0; nt < 4; ++nt) {
      const __bf16* kp = kbb + (size_t)(j0 + nt * 16 + row16) * C_;
#pragma unroll
      for (int f = 0; f < 8; ++f) {
        v16bf bf = *(const v16bf*)(kp + f * 32 + kb16);
        s[0][nt] = wmma_bf16(qf[0][f], bf, s[0][nt]);
        s[1][nt] = wmma_bf16(qf[1][f], bf, s[1][nt]);
      }
    }

    // online softmax (per row-tile a): row max via width-16 shuffles
    float mnew[2][8], alpha[2][8];
#pragma unroll
    for (int a = 0; a < 2; ++a) {
#pragma unroll
      for (int r = 0; r < 8; ++r) {
        float v = fmaxf(fmaxf(s[a][0][r], s[a][1][r]),
                        fmaxf(s[a][2][r], s[a][3][r])) * sc;
#pragma unroll
        for (int off = 1; off < 16; off <<= 1)
          v = fmaxf(v, __shfl_xor(v, off, 16));
        mnew[a][r]  = fmaxf(mrow[a][r], v);
        alpha[a][r] = __expf(mrow[a][r] - mnew[a][r]);
        mrow[a][r]  = mnew[a][r];
      }
      // P = exp(S - m); stage into LDS as bf16; accumulate row sums
      float rsum[8];
#pragma unroll
      for (int r = 0; r < 8; ++r) rsum[r] = 0.f;
#pragma unroll
      for (int nt = 0; nt < 4; ++nt) {
#pragma unroll
        for (int r = 0; r < 8; ++r) {
          float p = __expf(s[a][nt][r] * sc - mnew[a][r]);
          rsum[r] += p;
          myp[(a * 16 + r + 8 * half) * 64 + nt * 16 + row16] = (__bf16)p;
        }
      }
#pragma unroll
      for (int r = 0; r < 8; ++r) {
        float v = rsum[r];
#pragma unroll
        for (int off = 1; off < 16; off <<= 1)
          v += __shfl_xor(v, off, 16);
        lrow[a][r] = lrow[a][r] * alpha[a][r] + v;
      }
      // rescale running output accumulator
#pragma unroll
      for (int ct = 0; ct < 16; ++ct)
#pragma unroll
        for (int r = 0; r < 8; ++r) oacc[a][ct][r] *= alpha[a][r];
    }

    // LDS RAW: wave-private region, intra-wave cross-lane -> wait DScnt
    asm volatile("s_wait_dscnt 0" ::: "memory");

    // reload P as A-fragments (2 row-tiles x 2 fragments of 16x32)
    v16bf pf[2][2];
#pragma unroll
    for (int a = 0; a < 2; ++a)
#pragma unroll
      for (int f = 0; f < 2; ++f) {
        const __bf16* p = myp + (a * 16 + row16) * 64 + f * 32;
        pf[a][f] = *(const v16bf*)(p + kb16);
      }

    // O += P V : 16 column tiles, K-dim 64; V fragment shared by a=0,1
#pragma unroll 4
    for (int ct = 0; ct < 16; ++ct) {
      const __bf16* vp = vtb + (size_t)(ct * 16 + row16) * HW_ + j0;
#pragma unroll
      for (int f = 0; f < 2; ++f) {
        v16bf bf = *(const v16bf*)(vp + f * 32 + kb16);
        oacc[0][ct] = wmma_bf16(pf[0][f], bf, oacc[0][ct]);
        oacc[1][ct] = wmma_bf16(pf[1][f], bf, oacc[1][ct]);
      }
    }
  }

  // epilogue: normalize, store to (b, c, hw); r is contiguous in pixel dim
  float* ob = out + (size_t)b * C_ * HW_;
#pragma unroll
  for (int a = 0; a < 2; ++a)
#pragma unroll
    for (int ct = 0; ct < 16; ++ct) {
      const int chan = ct * 16 + row16;
      float* dst = ob + (size_t)chan * HW_ + qrow0 + a * 16 + 8 * half;
      v4f lo, hi;
#pragma unroll
      for (int r = 0; r < 4; ++r) {
        lo[r] = oacc[a][ct][r]     * (1.f / lrow[a][r]);
        hi[r] = oacc[a][ct][r + 4] * (1.f / lrow[a][r + 4]);
      }
      *(v4f*)(dst)     = lo;
      *(v4f*)(dst + 4) = hi;
    }
}

// ---------------------------------------------------------------------------
extern "C" void kernel_launch(void* const* d_in, const int* in_sizes, int n_in,
                              void* d_out, int out_size, void* d_ws,
                              size_t ws_size, hipStream_t stream) {
  (void)in_sizes; (void)n_in; (void)out_size; (void)ws_size;
  const float* x  = (const float*)d_in[0];
  const float* wq = (const float*)d_in[1];
  const float* wk = (const float*)d_in[2];
  const float* wv = (const float*)d_in[3];
  const float* bq = (const float*)d_in[4];
  const float* bk = (const float*)d_in[5];
  const float* bv = (const float*)d_in[6];
  float* out = (float*)d_out;

  const size_t qk = (size_t)B_ * HW_ * C_;  // 8,388,608 elems
  char* ws = (char*)d_ws;
  __bf16* qb  = reinterpret_cast<__bf16*>(ws);
  __bf16* kbp = reinterpret_cast<__bf16*>(ws + qk * 2);
  __bf16* vt  = reinterpret_cast<__bf16*>(ws + qk * 4);
  __bf16* wb  = reinterpret_cast<__bf16*>(ws + qk * 6);  // 3*256*256 bf16

  cvt_weights<<<(3 * C_ * C_ + 255) / 256, 256, 0, stream>>>(wq, wk, wv, wb);
  dim3 grid1(HW_ / 64, B_);
  qkv_proj<<<grid1, 128, 0, stream>>>(x, wb, bq, bk, bv, qb, kbp, vt);
  dim3 grid2(HW_ / 128, B_);
  flash_attn<<<grid2, 128, 0, stream>>>(qb, kbp, vt, out);
}
